// DynamicHexPlaneField_4183298146774
// MI455X (gfx1250) — compile-verified, matrix operand order checked by reading, not run on hardware
//
#include <hip/hip_runtime.h>
#include <hip/hip_fp16.h>
#include <math.h>

typedef __attribute__((ext_vector_type(16))) _Float16 v16h;
typedef __attribute__((ext_vector_type(8)))  _Float16 v8h;
typedef __attribute__((ext_vector_type(8)))  float    v8f;
typedef __attribute__((ext_vector_type(4)))  float    v4f;

#define CCH 32   // channels

// ---------------------------------------------------------------------------
// CDNA5 async global->LDS copy (16B per lane), ASYNCcnt-tracked.
// Builtin param0 type (from hipcc diagnostic): v4i32 pointer in AS(1).
// ---------------------------------------------------------------------------
typedef int v4i_vs __attribute__((vector_size(16)));
typedef __attribute__((address_space(1))) v4i_vs* async_gvp;
typedef __attribute__((address_space(3))) v4i_vs* async_lvp;

__device__ __forceinline__ void async_copy_b128(const _Float16* g, _Float16* l) {
#if __has_builtin(__builtin_amdgcn_global_load_async_to_lds_b128)
  async_gvp gp = (async_gvp)(__attribute__((address_space(1))) void*)(void*)g;
  async_lvp lp = (async_lvp)(__attribute__((address_space(3))) void*)(void*)l;
  __builtin_amdgcn_global_load_async_to_lds_b128(gp, lp, 0, 0);
#else
  unsigned lo = (unsigned)(size_t)(__attribute__((address_space(3))) void*)l;
  asm volatile("global_load_async_to_lds_b128 %0, %1, off"
               :: "v"(lo), "v"((unsigned long long)(size_t)g) : "memory");
#endif
}

__device__ __forceinline__ void wait_async_zero() {
#if __has_builtin(__builtin_amdgcn_s_wait_asynccnt)
  __builtin_amdgcn_s_wait_asynccnt(0);
#else
  asm volatile("s_wait_asynccnt 0x0" ::: "memory");
#endif
}

// ---------------------------------------------------------------------------
// Spectral norm (power iteration, 10 iters) + pack w/sigma into WMMA weight
// fragments (f16). Per-lane fragment layout (identical for A- or B-side use,
// per ISA 7.12.2 mirrored 16-bit layouts): lane L -> co' = L%16,
// element e -> ci = 16*(e>>3) + 8*(L>>4) + (e&7).
// Buffer index: ((tap*2 + cotile)*512 + lane*16 + e) halves per conv.
// ---------------------------------------------------------------------------
__global__ __launch_bounds__(288)
void specnorm_pack_kernel(const float* __restrict__ w0,
                          const float* __restrict__ w1,
                          _Float16* __restrict__ wfrag) {
  __shared__ float su[32];
  __shared__ float sv[288];
  __shared__ float red[288];
  __shared__ float s_scalar;
  __shared__ float s_sigma;
  const int tid = threadIdx.x;
  for (int conv = 0; conv < 2; ++conv) {
    const float* w = conv ? w1 : w0;   // [co=32][ci*9 + ky*3 + kx = 288]
    if (tid < 32) su[tid] = 0.17677669529663687f;  // 1/sqrt(32)
    __syncthreads();
    for (int it = 0; it < 10; ++it) {
      float vj = 0.f;
      for (int i = 0; i < 32; ++i) vj += w[i * 288 + tid] * su[i];
      red[tid] = vj * vj;
      __syncthreads();
      if (tid == 0) {
        float s = 0.f;
        for (int j = 0; j < 288; ++j) s += red[j];
        s_scalar = 1.0f / (sqrtf(s) + 1e-12f);
      }
      __syncthreads();
      sv[tid] = vj * s_scalar;
      __syncthreads();
      float ui = 0.f;
      if (tid < 32) {
        for (int j = 0; j < 288; ++j) ui += w[tid * 288 + j] * sv[j];
        red[tid] = ui * ui;
      }
      __syncthreads();
      if (tid == 0) {
        float s = 0.f;
        for (int i = 0; i < 32; ++i) s += red[i];
        float nrm = sqrtf(s);
        s_sigma  = nrm;                 // sigma = ||W v|| of final iterate
        s_scalar = 1.0f / (nrm + 1e-12f);
      }
      __syncthreads();
      if (tid < 32) su[tid] = ui * s_scalar;
      __syncthreads();
    }
    const float invsigma = 1.0f / s_sigma;
    for (int idx = tid; idx < 9216; idx += 288) {
      int t   = idx / 1024;
      int rem = idx - t * 1024;
      int n   = rem >> 9;
      int pos = rem & 511;
      int L   = pos >> 4;
      int e   = pos & 15;
      int co  = 16 * n + (L & 15);
      int ci  = 16 * (e >> 3) + 8 * (L >> 4) + (e & 7);
      wfrag[conv * 9216 + idx] = (_Float16)(w[co * 288 + ci * 9 + t] * invsigma);
    }
    __syncthreads();
  }
}

// ---------------------------------------------------------------------------
// 3x3 SAME conv, 32->32ch, implicit GEMM on v_wmma_f32_16x16x32_f16.
// Input: f16, zero-padded HWC layout [H+2][W+4][32] (interior (y,x) at
// row y+1, col x+2). Block = 256 thr (8 waves), block tile = 16x16 pixels,
// wave w computes rows 2w, 2w+1. Tile + packed weights staged to LDS with
// async global->LDS b128 copies.
// Tap loop is kx-major with a 4-row activation fragment cache: fragments for
// (rr, ky) depend only on tile row 2w+rr+ky in {2w..2w+3}, so 4 fragment
// loads serve all 6 (rr,ky) uses per kx (24 instead of 36 act ds_reads).
// MODE 0: D = W x Act, out = f16 HWC padded, relu(acc+bias)
//         (lane = pixel, vgpr = channel -> 16B contiguous f16 stores)
// MODE 1: D = Act x W, out = f32 CHW flat, scale*(acc+bias)
//         (lane = channel, vgpr = pixel -> b128 f32 stores)
// ---------------------------------------------------------------------------
template <int MODE>
__global__ __launch_bounds__(256)
void conv3x3_wmma_kernel(void* __restrict__ outp,
                         const _Float16* __restrict__ inh,
                         const _Float16* __restrict__ wfrag,  // 9216 halves
                         const float* __restrict__ bias,
                         const float* __restrict__ scale_ptr,
                         int H, int W) {
  __shared__ _Float16 lds_tile[18 * 18 * 32];  // [row][col][ci] halves
  __shared__ _Float16 lds_w[9216];
  const int tid = threadIdx.x;
  const int c0  = blockIdx.x * 16;
  const int r0  = blockIdx.y * 16;
  const int PW  = W + 4;

  // Async-stage input tile: 18 slabs of 18cols*32ch = 1152B contiguous each.
  for (int c = tid; c < 1296; c += 256) {
    int slab = c / 72, wc = c - slab * 72;
    const _Float16* g = inh + ((size_t)(r0 + slab) * PW + (c0 + 1)) * 32 + wc * 8;
    async_copy_b128(g, &lds_tile[c * 8]);
  }
  // Async-stage packed weights (shared by all 8 waves): 18,432B.
  for (int c = tid; c < 1152; c += 256)
    async_copy_b128(wfrag + c * 8, &lds_w[c * 8]);
  wait_async_zero();
  __syncthreads();

  const int wave = tid >> 5;
  const int lane = tid & 31;
  const int hs   = lane >> 4;
  const int nn   = lane & 15;
  v8f a00 = {}, a01 = {}, a10 = {}, a11 = {};

  union U { v16h v; v8h h[2]; };
#pragma unroll
  for (int kx = 0; kx < 3; ++kx) {
    // 4-row activation fragment cache for this kx column.
    U G[4];
#pragma unroll
    for (int j = 0; j < 4; ++j) {
      const int bi = ((2 * wave + j) * 18 + (nn + kx)) * 32 + 8 * hs;
      G[j].h[0] = *(const v8h*)&lds_tile[bi];
      G[j].h[1] = *(const v8h*)&lds_tile[bi + 16];
    }
#pragma unroll
    for (int ky = 0; ky < 3; ++ky) {
      const int t = ky * 3 + kx;
      U W0, W1;
      const _Float16* wb = &lds_w[(t * 2) * 512 + lane * 16];
      W0.h[0] = *(const v8h*)wb;
      W0.h[1] = *(const v8h*)(wb + 8);
      W1.h[0] = *(const v8h*)(wb + 512);
      W1.h[1] = *(const v8h*)(wb + 520);
      const v16h F0 = G[ky].v;      // output row 2w   (rr=0)
      const v16h F1 = G[ky + 1].v;  // output row 2w+1 (rr=1)
      if (MODE == 0) {
        a00 = __builtin_amdgcn_wmma_f32_16x16x32_f16(false, W0.v, false, F0, (short)0, a00, false, false);
        a01 = __builtin_amdgcn_wmma_f32_16x16x32_f16(false, W1.v, false, F0, (short)0, a01, false, false);
        a10 = __builtin_amdgcn_wmma_f32_16x16x32_f16(false, W0.v, false, F1, (short)0, a10, false, false);
        a11 = __builtin_amdgcn_wmma_f32_16x16x32_f16(false, W1.v, false, F1, (short)0, a11, false, false);
      } else {
        a00 = __builtin_amdgcn_wmma_f32_16x16x32_f16(false, F0, false, W0.v, (short)0, a00, false, false);
        a01 = __builtin_amdgcn_wmma_f32_16x16x32_f16(false, F0, false, W1.v, (short)0, a01, false, false);
        a10 = __builtin_amdgcn_wmma_f32_16x16x32_f16(false, F1, false, W0.v, (short)0, a10, false, false);
        a11 = __builtin_amdgcn_wmma_f32_16x16x32_f16(false, F1, false, W1.v, (short)0, a11, false, false);
      }
    }
  }

  if (MODE == 0) {
    // D[M=co'][N=pixel]: lane -> pixel nn, vgpr q -> co = 16n + 8hs + q.
    _Float16* out = (_Float16*)outp;
#pragma unroll
    for (int rr = 0; rr < 2; ++rr) {
      const int row = r0 + 2 * wave + rr;
      _Float16* rp = out + ((size_t)(row + 1) * PW + (c0 + nn + 2)) * 32;
#pragma unroll
      for (int n = 0; n < 2; ++n) {
        v8f acc = rr ? (n ? a11 : a10) : (n ? a01 : a00);
        v8h hv;
#pragma unroll
        for (int q = 0; q < 8; ++q) {
          float r = acc[q] + bias[16 * n + 8 * hs + q];
          hv[q] = (_Float16)fmaxf(r, 0.f);
        }
        *(v8h*)(rp + 16 * n + 8 * hs) = hv;   // 16B contiguous f16 store
      }
    }
  } else {
    // D[M=pixel][N=co']: lane -> co = 16n + nn, vgpr q -> pixel c0 + 8hs + q.
    float* out = (float*)outp;
    const float scale = scale_ptr[0];
    const int HW = H * W;
#pragma unroll
    for (int rr = 0; rr < 2; ++rr) {
      const int row = r0 + 2 * wave + rr;
#pragma unroll
      for (int n = 0; n < 2; ++n) {
        v8f acc = rr ? (n ? a11 : a10) : (n ? a01 : a00);
        const int co = 16 * n + nn;
        const float b = bias[co];
        float* op = out + (size_t)co * HW + (size_t)row * W + c0 + 8 * hs;
        v4f lo, hi;
#pragma unroll
        for (int q = 0; q < 4; ++q) lo[q] = (acc[q] + b) * scale;
#pragma unroll
        for (int q = 0; q < 4; ++q) hi[q] = (acc[4 + q] + b) * scale;
        *(v4f*)op       = lo;
        *(v4f*)(op + 4) = hi;
      }
    }
  }
}

// ---------------------------------------------------------------------------
// out = y + (t/6) * sum(c_i * k_i)   (all f32 flat CHW)
// ---------------------------------------------------------------------------
__global__ __launch_bounds__(256)
void combine_kernel(float* __restrict__ out, const float* __restrict__ y,
                    const float* __restrict__ k1, const float* __restrict__ k2,
                    const float* __restrict__ k3, const float* __restrict__ k4,
                    const float* __restrict__ k5, const float* __restrict__ k6,
                    float c1, float c2, float c3, float c4, float c5, float c6,
                    const float* __restrict__ time_ptr, int n) {
  const float dt = time_ptr[0] * (1.0f / 6.0f);
  int i = blockIdx.x * blockDim.x + threadIdx.x;
  int stride = gridDim.x * blockDim.x;
  for (; i < n; i += stride) {
    float s = 0.f;
    if (c1 != 0.f) s += c1 * k1[i];
    if (c2 != 0.f) s += c2 * k2[i];
    if (c3 != 0.f) s += c3 * k3[i];
    if (c4 != 0.f) s += c4 * k4[i];
    if (c5 != 0.f) s += c5 * k5[i];
    if (c6 != 0.f) s += c6 * k6[i];
    out[i] = y[i] + dt * s;
  }
}

// ---------------------------------------------------------------------------
// Same combine but writes f16 into the padded HWC buffer (interior only).
// One thread per pixel; per-ci reads are coalesced across threads; writes are
// 4x 16B contiguous per thread. With all c_i == 0 this is a cvt/transpose.
// ---------------------------------------------------------------------------
__global__ __launch_bounds__(256)
void combine_f16_kernel(_Float16* __restrict__ outh, const float* __restrict__ y,
                        const float* __restrict__ k1, const float* __restrict__ k2,
                        const float* __restrict__ k3, const float* __restrict__ k4,
                        const float* __restrict__ k5, const float* __restrict__ k6,
                        float c1, float c2, float c3, float c4, float c5, float c6,
                        const float* __restrict__ time_ptr, int H, int W) {
  const float dt = time_ptr[0] * (1.0f / 6.0f);
  const int HW = H * W, PW = W + 4;
  const int p = blockIdx.x * blockDim.x + threadIdx.x;
  if (p >= HW) return;
  const int yy = p / W, xx = p - yy * W;
  _Float16* op = outh + ((size_t)(yy + 1) * PW + xx + 2) * 32;
#pragma unroll
  for (int g = 0; g < 4; ++g) {
    v8h hv;
#pragma unroll
    for (int e = 0; e < 8; ++e) {
      const size_t idx = (size_t)(g * 8 + e) * HW + p;
      float s = 0.f;
      if (c1 != 0.f) s += c1 * k1[idx];
      if (c2 != 0.f) s += c2 * k2[idx];
      if (c3 != 0.f) s += c3 * k3[idx];
      if (c4 != 0.f) s += c4 * k4[idx];
      if (c5 != 0.f) s += c5 * k5[idx];
      if (c6 != 0.f) s += c6 * k6[idx];
      hv[e] = (_Float16)(y[idx] + dt * s);
    }
    *(v8h*)(op + g * 8) = hv;
  }
}

__global__ __launch_bounds__(256)
void copy_kernel(float* __restrict__ dst, const float* __restrict__ src, int n) {
  int i = blockIdx.x * blockDim.x + threadIdx.x;
  int stride = gridDim.x * blockDim.x;
  for (; i < n; i += stride) dst[i] = src[i];
}

__global__ __launch_bounds__(256)
void zero_u32_kernel(unsigned* __restrict__ dst, int n) {
  int i = blockIdx.x * blockDim.x + threadIdx.x;
  int stride = gridDim.x * blockDim.x;
  for (; i < n; i += stride) dst[i] = 0u;
}

// ---------------------------------------------------------------------------
extern "C" void kernel_launch(void* const* d_in, const int* in_sizes, int n_in,
                              void* d_out, int out_size, void* d_ws, size_t ws_size,
                              hipStream_t stream) {
  (void)in_sizes; (void)n_in; (void)out_size; (void)d_ws; (void)ws_size;
  const float* time  = (const float*)d_in[1];
  const float* g0    = (const float*)d_in[2];
  const float* g1    = (const float*)d_in[3];
  const float* w0    = (const float*)d_in[4];
  const float* b0    = (const float*)d_in[5];
  const float* w1    = (const float*)d_in[6];
  const float* b1    = (const float*)d_in[7];
  const float* scale = (const float*)d_in[8];
  float* out = (float*)d_out;

  const size_t FEAT_N = 32000000ull;          // 500000*64, zeroed at the end
  const size_t P1  = (size_t)CCH * 256 * 256; // 2,097,152
  const size_t P0  = (size_t)CCH * 128 * 128; //   524,288
  const size_t PH1 = (size_t)CCH * 258 * 260; // padded f16 elems (g1)
  const size_t PH0 = (size_t)CCH * 130 * 132; // padded f16 elems (g0)

  // Scratch layout inside the features region of d_out (floats):
  float* f32_1 = out;                                   // 8*P1
  _Float16* f16_1 = (_Float16*)(out + 8 * P1);          // 3*PH1 halves
  float* f32_0 = out + 8 * P1 + 3 * (PH1 / 2);          // 8*P0
  _Float16* f16_0 = (_Float16*)(f32_0 + 8 * P0);        // 3*PH0 halves
  _Float16* wfrag = (_Float16*)((float*)f16_0 + 3 * (PH0 / 2));  // 18432 halves
  float* g0_out = out + FEAT_N;
  float* g1_out = out + FEAT_N + P0;

  specnorm_pack_kernel<<<1, 288, 0, stream>>>(w0, w1, wfrag);

  const float AT[5][6] = {
    {0.2f, 0, 0, 0, 0, 0},
    {3.f/40.f, 9.f/40.f, 0, 0, 0, 0},
    {44.f/45.f, -56.f/15.f, 32.f/9.f, 0, 0, 0},
    {19372.f/6561.f, -25360.f/2187.f, 64448.f/6561.f, -212.f/729.f, 0, 0},
    {9017.f/3168.f, -355.f/33.f, 46732.f/5247.f, 49.f/176.f, -5103.f/18656.f, 0}};
  const float BT[6] = {35.f/384.f, 0.f, 500.f/1113.f, 125.f/192.f,
                       -2187.f/6784.f, 11.f/84.f};

  struct GCfg { float* F; _Float16* Hb; size_t PH; const float* gin; float* gout; int H, W; };
  GCfg cfgs[2] = { {f32_1, f16_1, PH1, g1, g1_out, 256, 256},
                   {f32_0, f16_0, PH0, g0, g0_out, 128, 128} };

  for (int gi = 0; gi < 2; ++gi) {
    GCfg& c = cfgs[gi];
    const int P  = CCH * c.H * c.W;
    const int HW = c.H * c.W;
    float* bufA = c.F;
    float* bufB = c.F + (size_t)P;
    float* K[6];
    for (int i = 0; i < 6; ++i) K[i] = c.F + (2 + i) * (size_t)P;
    _Float16* yh    = c.Hb;
    _Float16* ytmph = c.Hb + c.PH;
    _Float16* hh    = c.Hb + 2 * c.PH;

    const int eb  = (P + 255) / 256;
    const int ebp = (HW + 255) / 256;
    dim3 cgrid(c.W / 16, c.H / 16);

    // Clear the three padded f16 buffers (establishes the zero halos).
    zero_u32_kernel<<<(int)((3 * c.PH / 2 + 255) / 256), 256, 0, stream>>>(
        (unsigned*)c.Hb, (int)(3 * c.PH / 2));

    copy_kernel<<<eb, 256, 0, stream>>>(bufA, c.gin, P);
    // yh = f16(g_in) (all coefs zero -> pure convert/transpose)
    combine_f16_kernel<<<ebp, 256, 0, stream>>>(
        yh, c.gin, c.gin, c.gin, c.gin, c.gin, c.gin, c.gin,
        0.f, 0.f, 0.f, 0.f, 0.f, 0.f, time, c.H, c.W);

    for (int s = 0; s < 6; ++s) {
      float* y     = (s & 1) ? bufB : bufA;
      float* ynext = (s == 5) ? c.gout : ((s & 1) ? bufA : bufB);
      for (int j = 0; j < 6; ++j) {
        const _Float16* inh;
        if (j == 0) {
          inh = yh;
        } else {
          const float* a = AT[j - 1];
          combine_f16_kernel<<<ebp, 256, 0, stream>>>(
              ytmph, y, K[0], K[1], K[2], K[3], K[4], K[5],
              a[0], a[1], a[2], a[3], a[4], a[5], time, c.H, c.W);
          inh = ytmph;
        }
        conv3x3_wmma_kernel<0><<<cgrid, 256, 0, stream>>>(
            (void*)hh, inh, wfrag, b0, scale, c.H, c.W);
        conv3x3_wmma_kernel<1><<<cgrid, 256, 0, stream>>>(
            (void*)K[j], hh, wfrag + 9216, b1, scale, c.H, c.W);
      }
      combine_kernel<<<eb, 256, 0, stream>>>(
          ynext, y, K[0], K[1], K[2], K[3], K[4], K[5],
          BT[0], BT[1], BT[2], BT[3], BT[4], BT[5], time, P);
      if (s < 5) {
        combine_f16_kernel<<<ebp, 256, 0, stream>>>(
            yh, ynext, ynext, ynext, ynext, ynext, ynext, ynext,
            0.f, 0.f, 0.f, 0.f, 0.f, 0.f, time, c.H, c.W);
      }
    }
  }

  // Features output is zeros; also wipes all scratch inside d_out.
  zero_u32_kernel<<<4096, 256, 0, stream>>>((unsigned*)out, (int)FEAT_N);
}
